// HGT_GNN_90520730730792
// MI455X (gfx1250) — matrix operand order repaired and checked by reading, not compiled, not run on hardware
//
#include <hip/hip_runtime.h>
#include <hip/hip_bf16.h>
#include <math.h>

// ---------------------------------------------------------------------------
// HGT GNN forward for MI455X (gfx1250, wave32, WMMA).
//
// Dense linear layers use v_wmma_f32_16x16x32_f16 (f16 inputs, f32 accum):
// every GEMM here is [N x 64] x [64 x 64] -> bandwidth bound on 23.3 TB/s HBM
// (AI ~ 16 FLOP/B), so f16 conversion of operands costs nothing in time and
// WMMA keeps the VALU free for the edge gather/scatter phase.
//
// Weight matrix is staged into LDS in *fragment-major* (swizzled) order so
// each lane's 16-half B fragment is one contiguous 32-byte LDS read
// (2x ds_load_b128) instead of 16 strided ds_load_u16.
//
// Relation key/value transforms are folded into composite weights
//   kr = x @ (Wk[:,h*32:+32] @ A_h) + (bk_h @ A_h)
// so k/v are never materialized (saves ~400MB traffic + workspace).
// ---------------------------------------------------------------------------

typedef __attribute__((ext_vector_type(16))) _Float16 v16h;
typedef __attribute__((ext_vector_type(8)))  _Float16 v8h;
typedef __attribute__((ext_vector_type(8)))  float    v8f;

// ------------------------- utility kernels ---------------------------------

__global__ void k_zero(float* __restrict__ p, long n) {
    for (long i = (long)blockIdx.x * blockDim.x + threadIdx.x; i < n;
         i += (long)gridDim.x * blockDim.x)
        p[i] = 0.0f;
}

// ------------------------- WMMA GEMM: Y[N x 64] = X[N x Kin] @ W + b -------
// 256 threads = 8 waves; wave w -> (rowTile = w>>2, colTile = w&3):
// block covers 32 rows x 64 cols of Y. Kin in {32, 64} (K-loop of 32).
//
// LDS layouts:
//   Xs : [32 rows][64 halves]  (row-contiguous; A frag = 2x aligned v8h)
//   Wswz: fragment-major: element (kt, col, hh, kk) at
//         ((kt*64 + col)*2 + hh)*16 + kk   with K = kt*32 + hh*16 + kk
//         -> per-lane B frag = one aligned v16h (2x ds_load_b128)

__global__ __launch_bounds__(256) void k_wmma_lin64(
    const float* __restrict__ X, int nrows, int Kin,
    const float* __restrict__ W,      // [Kin x 64] row-major
    const float* __restrict__ bias,   // [64] or nullptr
    float* __restrict__ Y)            // [nrows x 64]
{
    __shared__ __attribute__((aligned(16))) _Float16 Xs[32 * 64];
    __shared__ __attribute__((aligned(32))) _Float16 Wswz[64 * 64];

    const int tid  = threadIdx.x;
    const int row0 = blockIdx.x * 32;

    // ---- stage weights, swizzled to fragment-major ------------------------
    // src index s = k*64 + c  ->  dst = ((kt*64 + c)*2 + hh)*16 + kk
    for (int s = tid * 4; s < Kin * 64; s += 256 * 4) {
        const float4 w4 = *(const float4*)&W[s];
        const float wv[4] = {w4.x, w4.y, w4.z, w4.w};
        #pragma unroll
        for (int u = 0; u < 4; ++u) {
            int si = s + u;
            int k = si >> 6, c = si & 63;
            int kt = k >> 5, hh = (k >> 4) & 1, kk = k & 15;
            Wswz[((kt * 64 + c) * 2 + hh) * 16 + kk] = (_Float16)wv[u];
        }
    }

    // ---- stage X tile (float4 global loads, convert, zero-pad tail) -------
    const int q = Kin >> 2;  // float4s per row (8 or 16)
    for (int idx = tid; idx < 32 * q; idx += 256) {
        int r = idx / q, c4 = idx % q;
        int gr = row0 + r;
        float4 v = make_float4(0.f, 0.f, 0.f, 0.f);
        if (gr < nrows) v = *(const float4*)&X[(long)gr * Kin + c4 * 4];
        _Float16* d = &Xs[r * 64 + c4 * 4];
        d[0] = (_Float16)v.x; d[1] = (_Float16)v.y;
        d[2] = (_Float16)v.z; d[3] = (_Float16)v.w;
    }
    // hint the next row tile into cache while we compute this one
    if (row0 + 32 < nrows)
        __builtin_prefetch(&X[(long)(row0 + 32) * Kin], 0, 1);
    __syncthreads();

    const int wave = tid >> 5, lane = tid & 31;
    const int colTile = wave & 3, rowTile = wave >> 2;
    const int m  = lane & 15;    // A row within tile / B,C column within tile
    const int hh = lane >> 4;    // lane half selector
    const int cn = colTile * 16 + m;

    v8f acc = {};
    const _Float16* xr = &Xs[(rowTile * 16 + m) * 64];

    for (int kb = 0; kb < Kin; kb += 32) {
        // A frag 16x32: halves 0..15 -> K = kb+hh*16 .. +15 (matches Wswz B frag)
        v8h alo = *(const v8h*)(xr + kb + hh * 16);
        v8h ahi = *(const v8h*)(xr + kb + hh * 16 + 8);
        v16h a;
        #pragma unroll
        for (int i = 0; i < 8; ++i) { a[i] = alo[i]; a[8 + i] = ahi[i]; }

        // B frag 32x16: one contiguous 32B chunk in the swizzled layout
        v16h b = *(const v16h*)&Wswz[(((kb >> 5) * 64 + cn) * 2 + hh) * 16];

        acc = __builtin_amdgcn_wmma_f32_16x16x32_f16(
            false, a, false, b, (short)0, acc, false, false);
    }

    const float bv = bias ? bias[cn] : 0.0f;
    // C/D layout: VGPR r -> M = r + 8*hh
    #pragma unroll
    for (int r = 0; r < 8; ++r) {
        int grow = row0 + rowTile * 16 + r + 8 * hh;
        if (grow < nrows)
            Y[(long)grow * 64 + cn] = acc[r] + bv;
    }
}

// --------- composite relation weight: CW = Wblk @ A_h, CB = bblk @ A_h -----

__global__ void k_compose(const float* __restrict__ W,    // [64 x 64]
                          const float* __restrict__ bvec, // [64]
                          const float* __restrict__ A,    // [2][32][32]
                          float* __restrict__ CW,         // [64 x 64]
                          float* __restrict__ CB)         // [64]
{
    int idx = blockIdx.x * 256 + threadIdx.x;
    if (idx >= 64 * 64) return;
    int i = idx >> 6, col = idx & 63;
    int h = col >> 5, e = col & 31;
    float s = 0.0f;
    #pragma unroll
    for (int d = 0; d < 32; ++d)
        s += W[i * 64 + h * 32 + d] * A[h * 1024 + d * 32 + e];
    CW[idx] = s;
    if (i == 0) {
        float sb = 0.0f;
        #pragma unroll
        for (int d = 0; d < 32; ++d)
            sb += bvec[h * 32 + d] * A[h * 1024 + d * 32 + e];
        CB[col] = sb;
    }
}

// ------------------------- edge phase --------------------------------------

__device__ __forceinline__ unsigned fenc(float f) {
    unsigned u = __float_as_uint(f);
    return (u >> 31) ? ~u : (u | 0x80000000u);
}
__device__ __forceinline__ float fdec(unsigned u) {
    return (u >> 31) ? __uint_as_float(u & 0x7FFFFFFFu) : __uint_as_float(~u);
}

__global__ void k_edge_score(const int* __restrict__ ei, int E,
                             const float* __restrict__ Q,   // dst type [N x 64]
                             const float* __restrict__ KR,  // src type [N x 64]
                             const float* __restrict__ p2,  // [2]
                             float* __restrict__ alpha,     // [E x 2]
                             unsigned* __restrict__ amaxb,  // [TOT x 2]
                             int dstOff)
{
    int e = blockIdx.x * 256 + threadIdx.x;
    if (e >= E) return;
    int src = ei[e], dst = ei[E + e];
    const float* qr = &Q[(long)dst * 64];
    const float* kr = &KR[(long)src * 64];
    const float scale = 0.17677669529663687f;  // 1/sqrt(32)
    #pragma unroll
    for (int h = 0; h < 2; ++h) {
        float s = 0.0f;
        #pragma unroll
        for (int d = 0; d < 32; ++d) s += qr[h * 32 + d] * kr[h * 32 + d];
        s *= p2[h] * scale;
        alpha[(long)e * 2 + h] = s;
        atomicMax(&amaxb[(long)(dst + dstOff) * 2 + h], fenc(s));
    }
}

__global__ void k_edge_exp(const int* __restrict__ ei, int E,
                           float* __restrict__ alpha,
                           const unsigned* __restrict__ amaxb,
                           float* __restrict__ den, int dstOff)
{
    int e = blockIdx.x * 256 + threadIdx.x;
    if (e >= E) return;
    int dst = ei[E + e];
    #pragma unroll
    for (int h = 0; h < 2; ++h) {
        float ex = expf(alpha[(long)e * 2 + h] -
                        fdec(amaxb[(long)(dst + dstOff) * 2 + h]));
        alpha[(long)e * 2 + h] = ex;
        atomicAdd(&den[(long)(dst + dstOff) * 2 + h], ex);
    }
}

__global__ void k_edge_msg(const int* __restrict__ ei, int E,
                           const float* __restrict__ alpha,  // exp'd
                           const float* __restrict__ den,
                           const float* __restrict__ VR,     // src type [N x 64]
                           float* __restrict__ msg,          // [TOT x 64]
                           int dstOff)
{
    long idx = (long)blockIdx.x * 256 + threadIdx.x;
    if (idx >= (long)E * 64) return;
    int e = (int)(idx >> 6), c = (int)(idx & 63), h = c >> 5;
    int src = ei[e], dst = ei[E + e];
    float w = alpha[(long)e * 2 + h] /
              (den[(long)(dst + dstOff) * 2 + h] + 1e-16f);
    atomicAdd(&msg[(long)(dst + dstOff) * 64 + c], VR[(long)src * 64 + c] * w);
}

// ------------------------- elementwise -------------------------------------

__global__ void k_gelu(float* __restrict__ p, long n) {
    for (long i = (long)blockIdx.x * 256 + threadIdx.x; i < n;
         i += (long)gridDim.x * 256) {
        float x = p[i];
        p[i] = 0.5f * x * (1.0f + erff(x * 0.70710678118654752f));
    }
}

__global__ void k_combine(float* __restrict__ X, const float* __restrict__ T,
                          const float* __restrict__ skipp, long n) {
    float sk = 1.0f / (1.0f + expf(-skipp[0]));
    for (long i = (long)blockIdx.x * 256 + threadIdx.x; i < n;
         i += (long)gridDim.x * 256)
        X[i] = fmaxf(sk * T[i] + (2.0f - sk) * X[i], 0.0f);  // sk*o+(1-sk)*x+x_old
}

// ------------------------- batchnorm ---------------------------------------

__global__ void k_colstats(const float* __restrict__ X, int nrows,
                           float* __restrict__ stat) {  // [sum64 | sumsq64]
    int c = threadIdx.x;  // 0..63
    float s = 0.0f, s2 = 0.0f;
    for (int r = blockIdx.x; r < nrows; r += gridDim.x) {
        float v = X[(long)r * 64 + c];
        s += v; s2 += v * v;
    }
    atomicAdd(&stat[c], s);
    atomicAdd(&stat[64 + c], s2);
}

__global__ void k_bn_apply(const float* __restrict__ T, float* __restrict__ X,
                           const float* __restrict__ stat,
                           const float* __restrict__ gamma,
                           const float* __restrict__ beta, int nrows, int relu) {
    long n = (long)nrows * 64;
    float inv = 1.0f / (float)nrows;
    for (long i = (long)blockIdx.x * 256 + threadIdx.x; i < n;
         i += (long)gridDim.x * 256) {
        int c = (int)(i & 63);
        float mu = stat[c] * inv;
        float var = stat[64 + c] * inv - mu * mu;
        float y = gamma[c] * (T[i] - mu) * rsqrtf(var + 1e-5f) + beta[c];
        X[i] = relu ? fmaxf(y, 0.0f) : y;
    }
}

// ------------------------- pooling + final MLP -----------------------------

__global__ void k_pool(const float* __restrict__ X, const int* __restrict__ batch,
                       float* __restrict__ pool, int colBase, int nrows) {
    long n = (long)nrows * 64;
    for (long i = (long)blockIdx.x * 256 + threadIdx.x; i < n;
         i += (long)gridDim.x * 256) {
        int r = (int)(i >> 6), c = (int)(i & 63);
        atomicAdd(&pool[(long)batch[r] * 128 + colBase + c], X[i]);
    }
}

__global__ void k_count(const int* __restrict__ batch, float* __restrict__ cnt,
                        int nrows) {
    for (long i = (long)blockIdx.x * 256 + threadIdx.x; i < nrows;
         i += (long)gridDim.x * 256)
        atomicAdd(&cnt[batch[i]], 1.0f);
}

__global__ void k_pool_div(float* __restrict__ pool, const float* __restrict__ cnt) {
    int idx = blockIdx.x * 256 + threadIdx.x;
    if (idx >= 32 * 128) return;
    int r = idx >> 7, c = idx & 127, t = c >> 6;
    pool[idx] /= fmaxf(cnt[t * 32 + r], 1.0f);
}

__global__ __launch_bounds__(256) void k_mlp(
    const float* __restrict__ pool,  // [32 x 128]
    const float* __restrict__ w0, const float* __restrict__ b0,  // 128x64
    const float* __restrict__ w1, const float* __restrict__ b1,  // 64x64
    const float* __restrict__ w2, const float* __restrict__ b2,  // 64x2
    float* __restrict__ out)         // [32 x 2]
{
    __shared__ float h0[32 * 128];
    __shared__ float h1[32 * 64];
    int tid = threadIdx.x;
    for (int i = tid; i < 32 * 128; i += 256) h0[i] = pool[i];
    __syncthreads();
    for (int i = tid; i < 32 * 64; i += 256) {
        int r = i >> 6, c = i & 63;
        float s = b0[c];
        for (int k = 0; k < 128; ++k) s += h0[r * 128 + k] * w0[k * 64 + c];
        h1[i] = fmaxf(s, 0.0f);
    }
    __syncthreads();
    for (int i = tid; i < 32 * 64; i += 256) {
        int r = i >> 6, c = i & 63;
        float s = b1[c];
        for (int k = 0; k < 64; ++k) s += h1[r * 64 + k] * w1[k * 64 + c];
        h0[i] = fmaxf(s, 0.0f);  // reuse h0 as [32x64]
    }
    __syncthreads();
    if (tid < 64) {
        int r = tid >> 1, c = tid & 1;
        float s = b2[c];
        for (int k = 0; k < 64; ++k) s += h0[r * 64 + k] * w2[k * 2 + c];
        out[tid] = s;
    }
}

// ------------------------- host driver -------------------------------------

extern "C" void kernel_launch(void* const* d_in, const int* in_sizes, int n_in,
                              void* d_out, int out_size, void* d_ws, size_t ws_size,
                              hipStream_t stream) {
    const int N   = in_sizes[0] / 32;  // 200000
    const int E   = in_sizes[2] / 2;   // 250000
    const int TOT = 2 * N;

    auto P = [&](int i) { return (const float*)d_in[i]; };
    const int* ei[4]    = {(const int*)d_in[2], (const int*)d_in[3],
                           (const int*)d_in[4], (const int*)d_in[5]};
    const int* batch[2] = {(const int*)d_in[6], (const int*)d_in[7]};

    // ---- params, jax pytree flatten order (dict keys sorted) --------------
    // top: convs < final < post < pre
    int pi = 8;
    struct ConvP {
        const float *alin_b[2], *alin_w[2];
        const float *kb[2], *kw[2], *qb[2], *qw[2], *vb[2], *vw[2];
        const float *ra[4], *rm[4], *rp[4];
        const float *skip[2];
    } cv[2];
    for (int c = 0; c < 2; ++c) {
        for (int t = 0; t < 2; ++t) { cv[c].alin_b[t] = P(pi++); cv[c].alin_w[t] = P(pi++); }
        for (int t = 0; t < 2; ++t) {
            cv[c].kb[t] = P(pi++); cv[c].kw[t] = P(pi++);
            cv[c].qb[t] = P(pi++); cv[c].qw[t] = P(pi++);
            cv[c].vb[t] = P(pi++); cv[c].vw[t] = P(pi++);
        }
        for (int r = 0; r < 4; ++r) { cv[c].ra[r] = P(pi++); cv[c].rm[r] = P(pi++); cv[c].rp[r] = P(pi++); }
        for (int t = 0; t < 2; ++t) cv[c].skip[t] = P(pi++);
    }
    const float *fin_b[3], *fin_w[3];
    for (int i = 0; i < 3; ++i) { fin_b[i] = P(pi++); fin_w[i] = P(pi++); }
    const float *po_beta[3][2], *po_gamma[3][2], *po_b[3][2], *po_w[3][2];
    for (int l = 0; l < 3; ++l)
        for (int t = 0; t < 2; ++t) {
            po_beta[l][t] = P(pi++); po_gamma[l][t] = P(pi++);
            po_b[l][t]    = P(pi++); po_w[l][t]     = P(pi++);
        }
    const float *pre_beta[2], *pre_gamma[2], *pre_b[2], *pre_w[2];
    for (int t = 0; t < 2; ++t) {
        pre_beta[t] = P(pi++); pre_gamma[t] = P(pi++);
        pre_b[t]    = P(pi++); pre_w[t]     = P(pi++);
    }

    // ---- workspace layout (~475 MB) ---------------------------------------
    char* wp = (char*)d_ws;
    auto alloc = [&](size_t bytes) {
        char* r = wp; wp += (bytes + 255) & ~(size_t)255; return (float*)r;
    };
    const size_t N64 = (size_t)N * 64;
    float* X[2] = {alloc(N64 * 4), alloc(N64 * 4)};
    float* T[2] = {alloc(N64 * 4), alloc(N64 * 4)};
    float* Q[2] = {alloc(N64 * 4), alloc(N64 * 4)};
    float* KRV  = alloc(N64 * 4);
    float* MSG  = alloc((size_t)TOT * 64 * 4);
    unsigned* AMAXB = (unsigned*)alloc((size_t)TOT * 2 * 4);
    float* DEN  = alloc((size_t)TOT * 2 * 4);
    float* ALPHA = alloc((size_t)4 * E * 2 * 4);
    float* STAT = alloc(128 * 4);
    float* CW   = alloc(64 * 64 * 4);
    float* CB   = alloc(64 * 4);
    float* POOL = alloc(32 * 128 * 4);
    float* CNT  = alloc(64 * 4);

    const int gN   = (N + 31) / 32;        // GEMM blocks (32 rows/block)
    const int gE   = (E + 255) / 256;
    const int gEM  = (int)(((long)E * 64 + 255) / 256);
    const int gGS  = 2048;                 // grid-stride elementwise

    // ---- pre: X_t = relu(bn(Xraw @ preW + preB)) --------------------------
    for (int t = 0; t < 2; ++t) {
        k_wmma_lin64<<<gN, 256, 0, stream>>>(P(t), N, 32, pre_w[t], pre_b[t], T[t]);
        k_zero<<<1, 256, 0, stream>>>(STAT, 128);
        k_colstats<<<1024, 64, 0, stream>>>(T[t], N, STAT);
        k_bn_apply<<<gGS, 256, 0, stream>>>(T[t], X[t], STAT, pre_gamma[t], pre_beta[t], N, 1);
    }

    // ---- 2x HGTConv -------------------------------------------------------
    for (int c = 0; c < 2; ++c) {
        for (int t = 0; t < 2; ++t)
            k_wmma_lin64<<<gN, 256, 0, stream>>>(X[t], N, 64, cv[c].qw[t], cv[c].qb[t], Q[t]);

        k_zero<<<gGS, 256, 0, stream>>>(MSG, (long)TOT * 64);
        k_zero<<<64, 256, 0, stream>>>((float*)AMAXB, (long)TOT * 2);  // 0 == ordered -inf
        k_zero<<<64, 256, 0, stream>>>(DEN, (long)TOT * 2);

        // pass 1: scores + segment max (kr via composite weight GEMM)
        for (int r = 0; r < 4; ++r) {
            int s = r >> 1, d = r & 1;
            k_compose<<<16, 256, 0, stream>>>(cv[c].kw[s], cv[c].kb[s], cv[c].ra[r], CW, CB);
            k_wmma_lin64<<<gN, 256, 0, stream>>>(X[s], N, 64, CW, CB, KRV);
            k_edge_score<<<gE, 256, 0, stream>>>(ei[r], E, Q[d], KRV, cv[c].rp[r],
                                                 ALPHA + (long)r * E * 2, AMAXB, d * N);
        }
        // pass 2: exp + segment sum
        for (int r = 0; r < 4; ++r)
            k_edge_exp<<<gE, 256, 0, stream>>>(ei[r], E, ALPHA + (long)r * E * 2,
                                               AMAXB, DEN, (r & 1) * N);
        // pass 3: vr via composite weight GEMM + weighted scatter
        for (int r = 0; r < 4; ++r) {
            int s = r >> 1, d = r & 1;
            k_compose<<<16, 256, 0, stream>>>(cv[c].vw[s], cv[c].vb[s], cv[c].rm[r], CW, CB);
            k_wmma_lin64<<<gN, 256, 0, stream>>>(X[s], N, 64, CW, CB, KRV);
            k_edge_msg<<<gEM, 256, 0, stream>>>(ei[r], E, ALPHA + (long)r * E * 2,
                                                DEN, KRV, MSG, d * N);
        }
        // node update: X = relu(sk * a_lin(gelu(msg)) + (2-sk) * X)
        for (int t = 0; t < 2; ++t) {
            float* m = MSG + (size_t)t * N64;
            k_gelu<<<gGS, 256, 0, stream>>>(m, (long)N64);
            k_wmma_lin64<<<gN, 256, 0, stream>>>(m, N, 64, cv[c].alin_w[t], cv[c].alin_b[t], T[t]);
            k_combine<<<gGS, 256, 0, stream>>>(X[t], T[t], cv[c].skip[t], (long)N64);
        }
    }

    // ---- 3x post: bn(lin(x)) (+relu except last) --------------------------
    for (int l = 0; l < 3; ++l)
        for (int t = 0; t < 2; ++t) {
            k_wmma_lin64<<<gN, 256, 0, stream>>>(X[t], N, 64, po_w[l][t], po_b[l][t], T[t]);
            k_zero<<<1, 256, 0, stream>>>(STAT, 128);
            k_colstats<<<1024, 64, 0, stream>>>(T[t], N, STAT);
            k_bn_apply<<<gGS, 256, 0, stream>>>(T[t], X[t], STAT, po_gamma[l][t],
                                                po_beta[l][t], N, l < 2 ? 1 : 0);
        }

    // ---- global mean pool + final MLP -------------------------------------
    k_zero<<<17, 256, 0, stream>>>(POOL, 32 * 128);
    k_zero<<<1, 256, 0, stream>>>(CNT, 64);
    for (int t = 0; t < 2; ++t) {
        k_pool<<<gGS, 256, 0, stream>>>(X[t], batch[t], POOL, t * 64, N);
        k_count<<<gGS, 256, 0, stream>>>(batch[t], CNT + t * 32, N);
    }
    k_pool_div<<<16, 256, 0, stream>>>(POOL, CNT);
    k_mlp<<<1, 256, 0, stream>>>(POOL, fin_w[0], fin_b[0], fin_w[1], fin_b[1],
                                 fin_w[2], fin_b[2], (float*)d_out);

    (void)n_in; (void)out_size; (void)ws_size;
}
